// DiffeqSolver_37426345017854
// MI455X (gfx1250) — matrix operand order repaired and checked by reading, not compiled, not run on hardware
//
#include <hip/hip_runtime.h>
#include <hip/hip_bf16.h>
#include <math.h>

typedef _Float16 h16;
typedef __attribute__((ext_vector_type(16))) _Float16 v16h;
typedef __attribute__((ext_vector_type(8)))  _Float16 v8h;
typedef __attribute__((ext_vector_type(8)))  float    v8f;

#define BB 32
#define CC 64
#define HH 28
#define WW 28
#define TT 40
#define MTOT (BB*HH*WW)          // 25088 output positions
#define NELEM (MTOT*CC)          // 1605632 state elements
#define KCHUNKS 18               // K = 9 taps * 64 ch = 576 = 18 * 32

// ---------------------------------------------------------------------------
// One-time: swizzle conv weights [O=64][I=64][3][3] f32 into WMMA B-fragment
// layout: [kchunk(18)][ntile(4)][lane(32)][elem(16)] f16.
// K index = tap*64 + ci. lane<16 holds k_local 0..15 (elem e = e),
// lane>=16 holds k_local 16..31 (elem e = 16+e). Column co = ntile*16 + lane%16.
// ---------------------------------------------------------------------------
__global__ void swizzle_weights_kernel(const float* __restrict__ Wsrc,
                                       h16* __restrict__ dst) {
    int r = blockIdx.x * 256 + threadIdx.x;
    if (r >= KCHUNKS * 4 * 32 * 16) return;
    int e    = r & 15;
    int lane = (r >> 4) & 31;
    int nt   = (r >> 9) & 3;
    int c    = r >> 11;
    int k    = c * 32 + ((lane & 16) ? 16 : 0) + e;
    int co   = nt * 16 + (lane & 15);
    int tap  = k >> 6;   // k / 64
    int ci   = k & 63;   // k % 64
    dst[r] = (h16)Wsrc[(co * 64 + ci) * 9 + tap];
}

// ---------------------------------------------------------------------------
// Entry: first_point NCHW f32 -> y NHWC f32, yh NHWC f16, d_out[:,0,...] copy.
// ---------------------------------------------------------------------------
__global__ void init_kernel(const float* __restrict__ fp,
                            float* __restrict__ y, h16* __restrict__ yh,
                            float* __restrict__ dout) {
    int idx = blockIdx.x * 256 + threadIdx.x;
    if (idx >= NELEM) return;
    int w   = idx % 28;
    int h   = (idx / 28) % 28;
    int cch = (idx / 784) % 64;
    int b   = idx / (784 * 64);
    float v = fp[idx];
    // d_out[b][0][c][h][w]
    dout[(((size_t)b * TT) * 64 + cch) * 784 + h * 28 + w] = v;
    size_t nhwc = ((size_t)(b * 28 + h) * 28 + w) * 64 + cch;
    y[nhwc]  = v;
    yh[nhwc] = (h16)v;
}

// ---------------------------------------------------------------------------
// Implicit-GEMM 3x3 SAME conv via v_wmma_f32_16x16x32_f16.
// x: NHWC f16 [32,28,28,64]; wsw: swizzled B-fragments; bias[64].
// Each wave: 16 rows (output positions) x 64 channels = 4 accumulators.
// Block = 4 waves (64 rows); grid = 25088/64 = 392 blocks.
// TANH16: write f16 tanh(out+bias); else write f32 out+bias.
// ---------------------------------------------------------------------------
template<bool TANH16>
__global__ __launch_bounds__(128)
void conv3x3_wmma_kernel(const h16* __restrict__ x,
                         const h16* __restrict__ wsw,
                         const float* __restrict__ bias,
                         float* __restrict__ outf,
                         h16* __restrict__ outh) {
    const int lane = threadIdx.x & 31;
    const int wave = threadIdx.x >> 5;
    const int m0   = blockIdx.x * 64 + wave * 16;

    // Row owned by this lane (A-matrix M = lane%16).
    const int row = m0 + (lane & 15);
    const int b   = row / 784;
    const int hw  = row - b * 784;
    const int h   = hw / 28;
    const int w   = hw - h * 28;
    const int klo = (lane & 16) ? 16 : 0;  // A layout: lanes>=16 hold K 16..31

    v8f acc[4] = {};

    const v16h* bfrag_base = (const v16h*)wsw;

#pragma unroll
    for (int tap = 0; tap < 9; ++tap) {
        const int hh  = h + tap / 3 - 1;
        const int ww2 = w + tap % 3 - 1;
        const bool inb = (hh >= 0) & (hh < 28) & (ww2 >= 0) & (ww2 < 28);
#pragma unroll
        for (int half = 0; half < 2; ++half) {
            const int c = tap * 2 + half;      // k-chunk index, k0 = 32*c
            v16h a = {};
            if (inb) {
                const long pix = ((long)b * 28 + hh) * 28 + ww2;
                const h16* p = x + pix * 64 + half * 32 + klo;
                v8h lo = *(const v8h*)p;         // K_local klo..klo+7
                v8h hi = *(const v8h*)(p + 8);   // K_local klo+8..klo+15
                a = __builtin_shufflevector(lo, hi,
                        0,1,2,3,4,5,6,7,8,9,10,11,12,13,14,15);
            }
            const v16h* bp = bfrag_base + (size_t)(c * 4) * 32 + lane;
#pragma unroll
            for (int t = 0; t < 4; ++t) {
                v16h bf = bp[t * 32];
                acc[t] = __builtin_amdgcn_wmma_f32_16x16x32_f16(
                    false, a, false, bf, (short)0, acc[t], false, false);
            }
        }
    }

    // Epilogue. C/D layout: VGPR j -> M = j (+8 for lanes>=16), N = lane%16.
    const int nn    = lane & 15;
    const int mbase = m0 + ((lane & 16) ? 8 : 0);
#pragma unroll
    for (int t = 0; t < 4; ++t) {
        const float bv = bias[t * 16 + nn];
#pragma unroll
        for (int j = 0; j < 8; ++j) {
            float v = acc[t][j] + bv;
            size_t off = (size_t)(mbase + j) * 64 + t * 16 + nn;
            if (TANH16) outh[off] = (h16)tanhf(v);
            else        outf[off] = v;
        }
    }
}

// ---------------------------------------------------------------------------
// RK4 (3/8 rule) elementwise combinations. dt read from the device time grid.
// MODE 1: o = y + (dt/3) k1
// MODE 2: o = y + dt k2 - (dt/3) k1
// MODE 3: o = y + dt (k1 - k2 + k3)
// MODE 4: o = y + dt (k1 + 3k2 + 3k3 + k4)/8   (also updates y f32, d_out)
// ---------------------------------------------------------------------------
template<int MODE>
__global__ void combo_kernel(const float* __restrict__ y,
                             const float* __restrict__ k1,
                             const float* __restrict__ k2,
                             const float* __restrict__ k3,
                             const float* __restrict__ k4,
                             const float* __restrict__ tgrid, int step,
                             float* __restrict__ outf, h16* __restrict__ outh,
                             float* __restrict__ dout, int tout) {
    int idx = blockIdx.x * 256 + threadIdx.x;
    if (idx >= NELEM) return;
    const float dt = tgrid[step + 1] - tgrid[step];
    float v;
    if (MODE == 1)      v = y[idx] + (dt * (1.0f / 3.0f)) * k1[idx];
    else if (MODE == 2) v = y[idx] + dt * k2[idx] - (dt * (1.0f / 3.0f)) * k1[idx];
    else if (MODE == 3) v = y[idx] + dt * (k1[idx] - k2[idx] + k3[idx]);
    else                v = y[idx] + (dt * 0.125f) *
                             (k1[idx] + 3.0f * k2[idx] + 3.0f * k3[idx] + k4[idx]);
    outh[idx] = (h16)v;
    if (MODE == 4) {
        outf[idx] = v;
        int cch = idx & 63;
        int m   = idx >> 6;
        int w   = m % 28;
        int h   = (m / 28) % 28;
        int b   = m / 784;
        dout[(((size_t)b * TT + tout) * 64 + cch) * 784 + h * 28 + w] = v;
    }
}

// ---------------------------------------------------------------------------
extern "C" void kernel_launch(void* const* d_in, const int* in_sizes, int n_in,
                              void* d_out, int out_size, void* d_ws, size_t ws_size,
                              hipStream_t stream) {
    const float* fp = (const float*)d_in[0];   // first_point [32,64,28,28]
    const float* ts = (const float*)d_in[1];   // time grid [40]
    const float* W1 = (const float*)d_in[2];
    const float* b1 = (const float*)d_in[3];
    const float* W2 = (const float*)d_in[4];
    const float* b2 = (const float*)d_in[5];
    float* out = (float*)d_out;

    char* ws = (char*)d_ws;
    const size_t SZ_F32 = (size_t)NELEM * 4;   // 6,422,528
    const size_t SZ_F16 = (size_t)NELEM * 2;   // 3,211,264
    float* y   = (float*)(ws);
    h16*   yh  = (h16*)(ws + SZ_F32);
    h16*   ah  = (h16*)(ws + SZ_F32 + SZ_F16);
    h16*   yth = (h16*)(ws + SZ_F32 + 2 * SZ_F16);
    float* k1  = (float*)(ws + SZ_F32 + 3 * SZ_F16);
    float* k2  = k1 + NELEM;
    float* k3  = k2 + NELEM;
    float* k4  = k3 + NELEM;
    h16*   w1s = (h16*)((char*)(k4 + NELEM));
    h16*   w2s = w1s + KCHUNKS * 4 * 32 * 16;

    const int CBLK = 392;   // 25088 rows / 64 per block
    const int EBLK = (NELEM + 255) / 256;

    swizzle_weights_kernel<<<(KCHUNKS*4*32*16 + 255)/256, 256, 0, stream>>>(W1, w1s);
    swizzle_weights_kernel<<<(KCHUNKS*4*32*16 + 255)/256, 256, 0, stream>>>(W2, w2s);
    init_kernel<<<EBLK, 256, 0, stream>>>(fp, y, yh, out);

    for (int i = 0; i < TT - 1; ++i) {
        // k1 = f(y)
        conv3x3_wmma_kernel<true ><<<CBLK, 128, 0, stream>>>(yh,  w1s, b1, nullptr, ah);
        conv3x3_wmma_kernel<false><<<CBLK, 128, 0, stream>>>(ah,  w2s, b2, k1, nullptr);
        combo_kernel<1><<<EBLK, 256, 0, stream>>>(y, k1, k1, k1, k1, ts, i,
                                                  nullptr, yth, nullptr, 0);
        // k2 = f(y + dt*k1/3)
        conv3x3_wmma_kernel<true ><<<CBLK, 128, 0, stream>>>(yth, w1s, b1, nullptr, ah);
        conv3x3_wmma_kernel<false><<<CBLK, 128, 0, stream>>>(ah,  w2s, b2, k2, nullptr);
        combo_kernel<2><<<EBLK, 256, 0, stream>>>(y, k1, k2, k2, k2, ts, i,
                                                  nullptr, yth, nullptr, 0);
        // k3 = f(y + dt*(k2 - k1/3))
        conv3x3_wmma_kernel<true ><<<CBLK, 128, 0, stream>>>(yth, w1s, b1, nullptr, ah);
        conv3x3_wmma_kernel<false><<<CBLK, 128, 0, stream>>>(ah,  w2s, b2, k3, nullptr);
        combo_kernel<3><<<EBLK, 256, 0, stream>>>(y, k1, k2, k3, k3, ts, i,
                                                  nullptr, yth, nullptr, 0);
        // k4 = f(y + dt*(k1 - k2 + k3))
        conv3x3_wmma_kernel<true ><<<CBLK, 128, 0, stream>>>(yth, w1s, b1, nullptr, ah);
        conv3x3_wmma_kernel<false><<<CBLK, 128, 0, stream>>>(ah,  w2s, b2, k4, nullptr);
        // y += dt*(k1 + 3k2 + 3k3 + k4)/8 ; store step i+1
        combo_kernel<4><<<EBLK, 256, 0, stream>>>(y, k1, k2, k3, k4, ts, i,
                                                  y, yh, out, i + 1);
    }
}
